// Discriminator_4930622455875
// MI455X (gfx1250) — matrix operand order repaired and checked by reading, not compiled, not run on hardware
//
#include <hip/hip_runtime.h>
#include <cstddef>
#include <cstdint>

// ---------------------------------------------------------------------------
// MI455X (gfx1250) implementation.
//   B=10000, TED=512, G=64, N=2000, E=64000, MD=15, NOISE=128
//   PAC=10, PACDIM=6400, D0=1024, D1=512
// Dominant work: [1000,6400]x[6400,1024] + [1000,1024]x[1024,512] GEMMs via
// v_wmma_f32_16x16x32_bf16 (wave32, f32 accumulate), with tiles staged
// global->LDS through the CDNA5 async path (GLOBAL_LOAD_ASYNC_TO_LDS_B128,
// ASYNCcnt). The per-sample noise matmul is factored through the 64 graphs.
// ---------------------------------------------------------------------------

typedef __attribute__((ext_vector_type(16))) __bf16 v16bf;
typedef __attribute__((ext_vector_type(8)))  __bf16 v8bf;
typedef __attribute__((ext_vector_type(8)))  float  v8f;

#define NG    64
#define NN    2000
#define NE    64000
#define NB    10000
#define NROW  1000
#define PACD  6400

// ---- CDNA5 async global->LDS helpers --------------------------------------
// LDS operand: low 32 bits of the generic pointer to a __shared__ object are
// the wave-relative LDS byte offset (addrspacecast local->flat yields
// {shared_aperture_hi32, lds_offset32}; hardware truncates to addr[31:0]).
__device__ __forceinline__ unsigned lds_off32(const void* p) {
  return (unsigned)(uintptr_t)p;
}
__device__ __forceinline__ void async_ld_b128(unsigned lds, const void* gaddr) {
  asm volatile("global_load_async_to_lds_b128 %0, %1, off"
               :: "v"(lds), "v"(gaddr) : "memory");
}
__device__ __forceinline__ void wait_async0() {
  asm volatile("s_wait_asynccnt 0x0" ::: "memory");
}

// ---------------------------------------------------------------------------
// GCN per graph (deterministic: integer degree atomics + Q32.32 fixed-point
// message accumulation in LDS so float rounding is order-independent).
// ---------------------------------------------------------------------------
__global__ __launch_bounds__(256) void gcn_kernel(
    const float* __restrict__ gx, const int* __restrict__ ei,
    const float* __restrict__ gcn_w, const float* __restrict__ gcn_b,
    float* __restrict__ gcn_out)
{
  __shared__ int degi[NN];
  __shared__ float xw[NN];
  __shared__ float dinv[NN];
  __shared__ unsigned long long msg[NN];
  const int g = blockIdx.x, t = threadIdx.x;
  const float w0 = gcn_w[0], b0 = gcn_b[0];
  for (int n = t; n < NN; n += 256) {
    degi[n] = 1;                       // self loop
    xw[n]   = gx[g * NN + n] * w0;
    msg[n]  = 0ull;
  }
  __syncthreads();
  const int* src = ei + (size_t)g * 2 * NE;
  const int* dst = src + NE;
  for (int e = t; e < NE; e += 256) atomicAdd(&degi[dst[e]], 1);   // exact
  __syncthreads();
  for (int n = t; n < NN; n += 256) dinv[n] = rsqrtf((float)degi[n]);
  __syncthreads();
  const float SC = 4294967296.0f;      // 2^32 fixed point
  for (int e = t; e < NE; e += 256) {
    int s = src[e], d = dst[e];
    float v = xw[s] * dinv[s] * dinv[d];
    atomicAdd(&msg[d], (unsigned long long)(long long)(v * SC));   // exact
  }
  __syncthreads();
  for (int n = t; n < NN; n += 256) {
    float m = (float)((double)(long long)msg[n] * (1.0 / 4294967296.0));
    gcn_out[g * NN + n] = m + xw[n] * dinv[n] * dinv[n] + b0;
  }
}

// ---------------------------------------------------------------------------
// gn[g][j] = sum_n gcn_out[g,n] * gme_w[n,j]   (g<64, j<128, n<2000)
// ---------------------------------------------------------------------------
__global__ __launch_bounds__(128) void gnoise_kernel(
    const float* __restrict__ gcn_out, const float* __restrict__ gme_w,
    float* __restrict__ gn)
{
  const int g = blockIdx.x, j = threadIdx.x;
  float acc = 0.f;
  for (int n = 0; n < NN; ++n)
    acc += gcn_out[g * NN + n] * gme_w[n * 128 + j];
  gn[g * 128 + j] = acc;
}

// ---------------------------------------------------------------------------
// Per-sample meta MLP + noise assembly + bf16 activation packing.
// One wave per sample b; row r=b/10, cols p*640+[0,512)=input_, [512,640)=noise.
// ---------------------------------------------------------------------------
__global__ __launch_bounds__(256) void noise_pack_kernel(
    const float* __restrict__ input_, const int* __restrict__ graph_ids,
    const float* __restrict__ chain, const float* __restrict__ metadata,
    const float* __restrict__ meta_w, const float* __restrict__ meta_b,
    const float* __restrict__ gme_w, const float* __restrict__ gme_b,
    const float* __restrict__ gn, __bf16* __restrict__ h)
{
  const int b = blockIdx.x * 8 + (threadIdx.x >> 5);
  const int lane = threadIdx.x & 31;
  float emb = meta_b[lane];
#pragma unroll
  for (int i = 0; i < 16; ++i) {
    float mi = (i == 0) ? chain[b] : metadata[b * 15 + (i - 1)];
    emb += mi * meta_w[i * 32 + lane];
  }
  emb = fmaxf(emb, 0.f);
  const int g = graph_ids[b];
  float nz[4];
#pragma unroll
  for (int q = 0; q < 4; ++q)
    nz[q] = gn[g * 128 + lane + 32 * q] + gme_b[lane + 32 * q];
  for (int o = 0; o < 32; ++o) {
    float v = __shfl(emb, o, 32);
#pragma unroll
    for (int q = 0; q < 4; ++q)
      nz[q] += v * gme_w[(NN + o) * 128 + lane + 32 * q];
  }
  const int r = b / 10, p = b % 10;
  __bf16* hrow = h + (size_t)r * PACD + p * 640;
#pragma unroll
  for (int c = lane; c < 512; c += 32)
    hrow[c] = (__bf16)input_[(size_t)b * 512 + c];
#pragma unroll
  for (int q = 0; q < 4; ++q)
    hrow[512 + lane + 32 * q] = (__bf16)nz[q];
}

// ---------------------------------------------------------------------------
// Tiled transpose + f32->bf16 convert:  Wt[n][k] = bf16(W[k][n]).
// ---------------------------------------------------------------------------
__global__ void transpose_to_bf16(const float* __restrict__ W,
                                  __bf16* __restrict__ Wt, int K, int N)
{
  __shared__ float tile[32][33];
  const int k0 = blockIdx.x * 32, n0 = blockIdx.y * 32;
  const int tx = threadIdx.x, ty = threadIdx.y;     // 32 x 8
#pragma unroll
  for (int i = ty; i < 32; i += 8)
    tile[i][tx] = W[(size_t)(k0 + i) * N + n0 + tx];
  __syncthreads();
#pragma unroll
  for (int i = ty; i < 32; i += 8)
    Wt[(size_t)(n0 + i) * K + k0 + tx] = (__bf16)tile[tx][i];
}

// ---------------------------------------------------------------------------
// WMMA GEMM:  C[M,N] = leaky_relu(A[M,K] * Bt[N,K]^T + bias, 0.2)  in bf16.
// Block tile 64(M) x 128(N), K step 64, double-buffered 48KB LDS staged with
// GLOBAL_LOAD_ASYNC_TO_LDS_B128 (ASYNCcnt). 8 waves, each a 16x64 strip =
// 4 f32 accumulators, 8 WMMAs per stage.
// A-tile rows are loaded without an M-bound: over-read lands in the adjacent
// (valid) workspace region and only feeds output rows masked in the epilogue.
// ---------------------------------------------------------------------------
__global__ __launch_bounds__(256) void gemm_bf16_wmma(
    const __bf16* __restrict__ A, const __bf16* __restrict__ Bt,
    const float* __restrict__ bias, __bf16* __restrict__ C,
    int M, int N, int K)
{
  __shared__ __bf16 As[2][64 * 64];    // 8KB per buffer
  __shared__ __bf16 Bs[2][128 * 64];   // 16KB per buffer
  const int tid  = threadIdx.x;
  const int lane = tid & 31;
  const int wave = tid >> 5;
  const int wm = wave & 3;          // M sub-tile (4 x 16)
  const int wn = wave >> 2;         // N sub-strip (2 x 64)
  const int m0 = blockIdx.x * 64;
  const int n0 = blockIdx.y * 128;
  const int larow   = lane & 15;
  const bool hiHalf = lane >= 16;

  v8f acc[4] = {};

  // staging geometry (per thread): A 64x64 halfs -> 2 x b128; B 128x64 -> 4.
  const int arow = tid >> 2;            // 0..63
  const int acol = (tid & 3) * 16;      // half offset 0,16,32,48
  const int brow = tid >> 1;            // 0..127
  const int bcol = (tid & 1) * 32;      // half offset 0,32

  const int nsteps = K >> 6;

  auto stage_async = [&](int s, int buf) {
    const int k0 = s << 6;
    const __bf16* ag = A + (size_t)(m0 + arow) * K + k0 + acol;
#pragma unroll
    for (int j = 0; j < 2; ++j)
      async_ld_b128(lds_off32(&As[buf][arow * 64 + acol + 8 * j]), ag + 8 * j);
    const __bf16* bg = Bt + (size_t)(n0 + brow) * K + k0 + bcol;
#pragma unroll
    for (int j = 0; j < 4; ++j)
      async_ld_b128(lds_off32(&Bs[buf][brow * 64 + bcol + 8 * j]), bg + 8 * j);
  };

  stage_async(0, 0);
  for (int s = 0; s < nsteps; ++s) {
    wait_async0();        // own async writes into buf (s&1) complete
    __syncthreads();      // everyone's writes visible; prior reads retired
    const int buf = s & 1;
    if (s + 1 < nsteps) stage_async(s + 1, buf ^ 1);

    const __bf16* at = As[buf];
    const __bf16* bt = Bs[buf];
    // CDNA5 16-bit fragment layouts: A and B lane<16 K{0..7,16..23}, lane>=16
    // K{8..15,24..31}.
    const int kbA = hiHalf ? 8 : 0;
    const int kbB = hiHalf ? 8 : 0;
#pragma unroll
    for (int u = 0; u < 2; ++u) {       // two k-substeps of 32
      const __bf16* ap = &at[(wm * 16 + larow) * 64 + u * 32 + kbA];
      v8bf a0 = *reinterpret_cast<const v8bf*>(ap);
      v8bf a1 = *reinterpret_cast<const v8bf*>(ap + 16);
      v16bf a = __builtin_shufflevector(a0, a1, 0, 1, 2, 3, 4, 5, 6, 7,
                                        8, 9, 10, 11, 12, 13, 14, 15);
      v16bf bfr[4];
#pragma unroll
      for (int f = 0; f < 4; ++f) {     // hoist all B loads ahead of WMMAs
        const __bf16* bp = &bt[(wn * 64 + f * 16 + larow) * 64 + u * 32 + kbB];
        v8bf b0 = *reinterpret_cast<const v8bf*>(bp);
        v8bf b1 = *reinterpret_cast<const v8bf*>(bp + 16);
        bfr[f] = __builtin_shufflevector(b0, b1, 0, 1, 2, 3, 4, 5, 6, 7,
                                         8, 9, 10, 11, 12, 13, 14, 15);
      }
#pragma unroll
      for (int f = 0; f < 4; ++f)
        acc[f] = __builtin_amdgcn_wmma_f32_16x16x32_bf16(
            false, a, false, bfr[f], (short)0, acc[f], false, false);
    }
  }

  // Epilogue: bias + leaky_relu(0.2), store bf16.
  // C/D layout: VGPR v holds row v (lanes 0-15) / row v+8 (lanes 16-31).
#pragma unroll
  for (int f = 0; f < 4; ++f) {
    const int col = n0 + wn * 64 + f * 16 + larow;
    const float bv = bias[col];
#pragma unroll
    for (int v = 0; v < 8; ++v) {
      const int row = m0 + wm * 16 + v + (hiHalf ? 8 : 0);
      if (row < M) {
        float x = acc[f][v] + bv;
        x = x > 0.f ? x : 0.2f * x;
        C[(size_t)row * N + col] = (__bf16)x;
      }
    }
  }
}

// ---------------------------------------------------------------------------
// Final layer: out[r] = h2[r,:512] . w2 + b2   (wave per row)
// ---------------------------------------------------------------------------
__global__ __launch_bounds__(256) void final_kernel(
    const __bf16* __restrict__ h2, const float* __restrict__ w2,
    const float* __restrict__ b2, float* __restrict__ out)
{
  const int r = blockIdx.x * 8 + (threadIdx.x >> 5);
  const int lane = threadIdx.x & 31;
  float acc = 0.f;
#pragma unroll
  for (int j = lane; j < 512; j += 32)
    acc += (float)h2[(size_t)r * 512 + j] * w2[j];
#pragma unroll
  for (int off = 16; off; off >>= 1) acc += __shfl_down(acc, off, 32);
  if (lane == 0) out[r] = acc + b2[0];
}

// ---------------------------------------------------------------------------
extern "C" void kernel_launch(void* const* d_in, const int* in_sizes, int n_in,
                              void* d_out, int out_size, void* d_ws, size_t ws_size,
                              hipStream_t stream)
{
  const float* input_     = (const float*)d_in[0];
  const float* graphs_x   = (const float*)d_in[1];
  const int*   edge_index = (const int*)d_in[2];
  const int*   graph_ids  = (const int*)d_in[3];
  const float* chain      = (const float*)d_in[4];
  const float* metadata   = (const float*)d_in[5];
  const float* gcn_w      = (const float*)d_in[6];
  const float* gcn_b      = (const float*)d_in[7];
  const float* meta_w     = (const float*)d_in[8];
  const float* meta_b     = (const float*)d_in[9];
  const float* gme_w      = (const float*)d_in[10];
  const float* gme_b      = (const float*)d_in[11];
  const float* seq_w0     = (const float*)d_in[12];
  const float* seq_b0     = (const float*)d_in[13];
  const float* seq_w1     = (const float*)d_in[14];
  const float* seq_b1     = (const float*)d_in[15];
  const float* seq_w2     = (const float*)d_in[16];
  const float* seq_b2     = (const float*)d_in[17];

  char* ws = (char*)d_ws;
  // 256B-aligned workspace carve-up (~30.6 MB total)
  float*  gcn_out = (float*)(ws + 0);           //  64*2000*4   = 512000
  float*  gn      = (float*)(ws + 512000);      //  64*128*4    = 32768
  __bf16* h       = (__bf16*)(ws + 544768);     //  1000*6400*2 = 12800000
  __bf16* Wt0     = (__bf16*)(ws + 13344768);   //  1024*6400*2 = 13107200
  __bf16* Wt1     = (__bf16*)(ws + 26451968);   //  512*1024*2  = 1048576
  __bf16* h1      = (__bf16*)(ws + 27500544);   //  1000*1024*2 = 2048000
  __bf16* h2      = (__bf16*)(ws + 29548544);   //  1000*512*2  = 1024000
  float*  out     = (float*)d_out;

  gcn_kernel<<<NG, 256, 0, stream>>>(graphs_x, edge_index, gcn_w, gcn_b, gcn_out);
  gnoise_kernel<<<NG, 128, 0, stream>>>(gcn_out, gme_w, gn);
  noise_pack_kernel<<<NB / 8, 256, 0, stream>>>(input_, graph_ids, chain, metadata,
                                                meta_w, meta_b, gme_w, gme_b, gn, h);
  transpose_to_bf16<<<dim3(200, 32), dim3(32, 8), 0, stream>>>(seq_w0, Wt0, 6400, 1024);
  transpose_to_bf16<<<dim3(32, 16), dim3(32, 8), 0, stream>>>(seq_w1, Wt1, 1024, 512);
  gemm_bf16_wmma<<<dim3(16, 8), 256, 0, stream>>>(h, Wt0, seq_b0, h1, NROW, 1024, 6400);
  gemm_bf16_wmma<<<dim3(16, 4), 256, 0, stream>>>(h1, Wt1, seq_b1, h2, NROW, 512, 1024);
  final_kernel<<<NROW / 8, 256, 0, stream>>>(h2, seq_w2, seq_b2, out);
}
